// PraxisAttention_45019847196787
// MI455X (gfx1250) — compile-verified, matrix operand order checked by reading, not run on hardware
//
#include <hip/hip_runtime.h>
#include <stdint.h>

typedef __attribute__((ext_vector_type(16))) _Float16 v16h;
typedef __attribute__((ext_vector_type(8)))  float    v8f;
typedef __attribute__((ext_vector_type(4)))  uint32_t u32x4;

union FragH { v16h h; u32x4 q[2]; };
union PackH { _Float16 h[2]; uint32_t u; };

#define WMMA_F16(a, b, c) \
  __builtin_amdgcn_wmma_f32_16x16x32_f16(false, (a), false, (b), (short)0, (c), false, false)

__device__ __forceinline__ v8f vzero8() {
  v8f z = {0.f, 0.f, 0.f, 0.f, 0.f, 0.f, 0.f, 0.f};
  return z;
}

// A-fragment (16x32 f16, M x K) from row-major f16 matrix. Per ISA 16-bit A layout:
// lane m = lane%16 is the row; half=lane/16 selects K ranges {half*8..+7, 16+half*8..+7}.
__device__ __forceinline__ v16h ldA(const _Float16* A, int row, int ld, int k0) {
  const int lane = threadIdx.x & 31;
  const _Float16* p = A + (size_t)(row + (lane & 15)) * ld + k0 + ((lane >> 4) << 3);
  FragH f;
  f.q[0] = *(const u32x4*)(p);
  f.q[1] = *(const u32x4*)(p + 16);
  return f.h;
}

// B-fragment (32x16 f16, K x N) from a packed dword matrix Bp[n][k/2] where each dword
// holds the (k, k+1) pair for column n. Mirrors the A layout with lane%16 = N.
__device__ __forceinline__ v16h ldBp(const uint32_t* Bp, int col, int ldp, int k0) {
  const int lane = threadIdx.x & 31;
  const uint32_t* p = Bp + (size_t)(col + (lane & 15)) * ldp + (k0 >> 1) + ((lane >> 4) << 2);
  FragH f;
  f.q[0] = *(const u32x4*)(p);
  f.q[1] = *(const u32x4*)(p + 8);
  return f.h;
}

// ---------------- elementwise / packing kernels ----------------

__global__ void k_cast_x(const float* __restrict__ src, _Float16* __restrict__ dst, int n) {
  int i = blockIdx.x * blockDim.x + threadIdx.x;
  if (i < n) dst[i] = (_Float16)src[i];
}

// W is [K=1024][N=1024] f32 row-major. Pack to Wp[n][kp] dwords = {W[2kp][n], W[2kp+1][n]}.
__global__ void k_pack_w(const float* __restrict__ W, uint32_t* __restrict__ Wp) {
  int i = blockIdx.x * blockDim.x + threadIdx.x;  // 1024 * 512 threads
  int kp = i & 511, n = i >> 9;
  PackH p;
  p.h[0] = (_Float16)W[(size_t)(2 * kp) * 1024 + n];
  p.h[1] = (_Float16)W[(size_t)(2 * kp + 1) * 1024 + n];
  Wp[(size_t)n * 512 + kp] = p.u;
}

// src: [B,S,1024] f32 (head h at cols h*64..). dst: [B,H,S,64] f16 with RoPE applied.
__global__ void k_rope_heads(const float* __restrict__ src, _Float16* __restrict__ dst) {
  int i = blockIdx.x * blockDim.x + threadIdx.x;  // B*H*S*32
  int d2 = i & 31;
  int s  = (i >> 5) & 2047;
  int h  = (i >> 16) & 15;
  int b  = i >> 20;
  const float* row = src + ((size_t)(b * 2048 + s)) * 1024 + h * 64;
  float t1 = row[d2], t2 = row[d2 + 32];
  // inv_freq = 10000^(-(2*d2)/64)
  float inv = __expf(-(float)(2 * d2) * (9.210340371976184f / 64.f));
  float fr = (float)s * inv;
  float sn, cs;
  __sincosf(fr, &sn, &cs);
  _Float16* orow = dst + ((size_t)((b * 16 + h) * 2048 + s)) * 64;
  orow[d2]      = (_Float16)(t1 * cs - t2 * sn);
  orow[d2 + 32] = (_Float16)(t1 * sn + t2 * cs);
}

// src: [B,S,1024] f32. dst: vt[B,H,64,S/2] dwords = {V[2sp][d], V[2sp+1][d]} per head.
__global__ void k_v_heads_t(const float* __restrict__ src, uint32_t* __restrict__ dst) {
  int i = blockIdx.x * blockDim.x + threadIdx.x;  // B*H*64*1024
  int sp = i & 1023;
  int d  = (i >> 10) & 63;
  int h  = (i >> 16) & 15;
  int b  = i >> 20;
  const float* col = src + ((size_t)(b * 2048 + 2 * sp)) * 1024 + h * 64 + d;
  PackH p;
  p.h[0] = (_Float16)col[0];
  p.h[1] = (_Float16)col[1024];
  dst[((size_t)((b * 16 + h) * 64 + d)) * 1024 + sp] = p.u;
}

// ---------------- WMMA GEMM: C[f32, MxN] = A[f16, MxK] * packed-B ----------------
// Block = 256 threads = 8 waves; wave tile 32x64 (2x4 WMMA tiles, 8 WMMAs / 12 b128
// loads per k-step); block tile 128x128 (4 row-waves x 2 col-waves).
__global__ __launch_bounds__(256) void k_gemm(const _Float16* __restrict__ A,
                                              const uint32_t* __restrict__ Bp,
                                              float* __restrict__ C,
                                              int N, int K) {
  const int wid = threadIdx.x >> 5;
  const int lane = threadIdx.x & 31;
  const int row0 = blockIdx.y * 128 + (wid & 3) * 32;
  const int col0 = blockIdx.x * 128 + (wid >> 2) * 64;
  const int cn = lane & 15, hc = lane >> 4;
  const int ldp = K >> 1;
  v8f acc[2][4];
#pragma unroll
  for (int i = 0; i < 2; i++)
#pragma unroll
    for (int j = 0; j < 4; j++) acc[i][j] = vzero8();

  for (int k0 = 0; k0 < K; k0 += 32) {
    v16h fa[2];
    fa[0] = ldA(A, row0, K, k0);
    fa[1] = ldA(A, row0 + 16, K, k0);
    v16h fb[4];
#pragma unroll
    for (int j = 0; j < 4; j++) fb[j] = ldBp(Bp, col0 + j * 16, ldp, k0);
#pragma unroll
    for (int i = 0; i < 2; i++)
#pragma unroll
      for (int j = 0; j < 4; j++) acc[i][j] = WMMA_F16(fa[i], fb[j], acc[i][j]);
  }
#pragma unroll
  for (int i = 0; i < 2; i++) {
#pragma unroll
    for (int r = 0; r < 8; r++) {
      const int row = row0 + i * 16 + r + 8 * hc;
      float* crow = C + (size_t)row * N + col0 + cn;
#pragma unroll
      for (int j = 0; j < 4; j++) crow[j * 16] = acc[i][j][r];
    }
  }
}

// ---------------- Flash-style causal attention ----------------
// One wave per (b, h, 16-query tile). Block = 8 waves, each with a private 1KB LDS slice
// used to re-layout softmax probs into the A-fragment for the PV WMMA.
__global__ __launch_bounds__(256) void k_attn(const _Float16* __restrict__ qh,
                                              const _Float16* __restrict__ kh,
                                              const uint32_t* __restrict__ vt,
                                              const float* __restrict__ amask,
                                              _Float16* __restrict__ attnh) {
  __shared__ _Float16 lds[8 * 512];
  const int wid = threadIdx.x >> 5, lane = threadIdx.x & 31;
  const int gw = blockIdx.x * 8 + wid;
  const int qt = gw & 127;          // S/16 = 128 query tiles
  const int h  = (gw >> 7) & 15;
  const int b  = gw >> 11;
  const int q0 = qt << 4;
  const int cn = lane & 15, hc = lane >> 4;
  const _Float16* qhead = qh + ((size_t)(b * 16 + h)) * 2048 * 64;
  const uint32_t* khead = (const uint32_t*)(kh + ((size_t)(b * 16 + h)) * 2048 * 64);
  const uint32_t* vhead = vt + ((size_t)(b * 16 + h)) * 64 * 1024;
  _Float16* lsl = &lds[wid * 512];

  v16h qa0 = ldA(qhead, q0, 64, 0);
  v16h qa1 = ldA(qhead, q0, 64, 32);

  v8f accO[4];
#pragma unroll
  for (int j = 0; j < 4; j++) accO[j] = vzero8();
  float mrun[8], lrun[8];
#pragma unroll
  for (int r = 0; r < 8; r++) { mrun[r] = -3.0e38f; lrun[r] = 0.f; }

  const int kend = ((q0 + 16) + 31) & ~31;  // causal: only tiles up to diagonal
  for (int kb = 0; kb < kend; kb += 32) {
    v8f sc[2];
#pragma unroll
    for (int n = 0; n < 2; n++) {
      v16h kf0 = ldBp(khead, kb + n * 16, 32, 0);
      v16h kf1 = ldBp(khead, kb + n * 16, 32, 32);
      v8f c = vzero8();
      c = WMMA_F16(qa0, kf0, c);   // d = 0..31
      c = WMMA_F16(qa1, kf1, c);   // d = 32..63
      sc[n] = c;
    }
    float am0 = (1.0f - amask[b * 2048 + kb + cn]) * -1.0e9f;
    float am1 = (1.0f - amask[b * 2048 + kb + 16 + cn]) * -1.0e9f;
#pragma unroll
    for (int r = 0; r < 8; r++) {
      const int sq = q0 + r + 8 * hc;
      float s0 = sc[0][r] * 0.125f + am0;   // 1/sqrt(64)
      float s1 = sc[1][r] * 0.125f + am1;
      if (kb + cn > sq)      s0 += -1.0e9f;
      if (kb + 16 + cn > sq) s1 += -1.0e9f;
      // row max across the 16 lanes of this half (wave32 halves stay disjoint for masks<=8)
      float t = fmaxf(s0, s1);
      t = fmaxf(t, __shfl_xor(t, 1, 32));
      t = fmaxf(t, __shfl_xor(t, 2, 32));
      t = fmaxf(t, __shfl_xor(t, 4, 32));
      t = fmaxf(t, __shfl_xor(t, 8, 32));
      const float mo = mrun[r];
      const float mn = fmaxf(mo, t);
      const float sca = __expf(mo - mn);
      const float p0 = __expf(s0 - mn);
      const float p1 = __expf(s1 - mn);
      float rs = p0 + p1;
      rs += __shfl_xor(rs, 1, 32);
      rs += __shfl_xor(rs, 2, 32);
      rs += __shfl_xor(rs, 4, 32);
      rs += __shfl_xor(rs, 8, 32);
      mrun[r] = mn;
      lrun[r] = lrun[r] * sca + rs;
#pragma unroll
      for (int j = 0; j < 4; j++) accO[j][r] *= sca;
      lsl[(r + 8 * hc) * 32 + cn]      = (_Float16)p0;
      lsl[(r + 8 * hc) * 32 + 16 + cn] = (_Float16)p1;
    }
    asm volatile("s_wait_dscnt 0" ::: "memory");  // private slice: wave-local ordering only
    // reload P tile (16x32) from LDS in A-fragment layout
    FragH pf;
    const _Float16* pb = lsl + cn * 32 + hc * 8;
    pf.q[0] = *(const u32x4*)(pb);
    pf.q[1] = *(const u32x4*)(pb + 16);
#pragma unroll
    for (int j = 0; j < 4; j++) {
      v16h vf = ldBp(vhead + (kb >> 1), j * 16, 1024, 0);
      accO[j] = WMMA_F16(pf.h, vf, accO[j]);
    }
  }
#pragma unroll
  for (int r = 0; r < 8; r++) {
    const float inv = 1.0f / lrun[r];
    const int row = q0 + r + 8 * hc;
    _Float16* orow = attnh + ((size_t)(b * 2048 + row)) * 1024 + h * 64;
#pragma unroll
    for (int j = 0; j < 4; j++)
      orow[j * 16 + cn] = (_Float16)(accO[j][r] * inv);
  }
}

// ---------------- launcher ----------------

extern "C" void kernel_launch(void* const* d_in, const int* in_sizes, int n_in,
                              void* d_out, int out_size, void* d_ws, size_t ws_size,
                              hipStream_t stream) {
  (void)in_sizes; (void)n_in; (void)out_size; (void)ws_size;
  const float* x  = (const float*)d_in[0];
  const float* am = (const float*)d_in[1];
  const float* Wq = (const float*)d_in[2];
  const float* Wk = (const float*)d_in[3];
  const float* Wv = (const float*)d_in[4];
  const float* Wo = (const float*)d_in[5];
  float* out = (float*)d_out;
  char* ws = (char*)d_ws;
  const size_t MB = (size_t)1 << 20;
  _Float16* xh    = (_Float16*)(ws);             //  8 MB: x as f16 [4096,1024]
  uint32_t* wpq   = (uint32_t*)(ws + 8 * MB);    //  2 MB each: packed weights
  uint32_t* wpk   = (uint32_t*)(ws + 10 * MB);
  uint32_t* wpv   = (uint32_t*)(ws + 12 * MB);
  uint32_t* wpo   = (uint32_t*)(ws + 14 * MB);
  float*    pf32  = (float*)(ws + 16 * MB);      // 16 MB: reused f32 projection buffer
  _Float16* qhp   = (_Float16*)(ws + 32 * MB);   //  8 MB: Q heads f16 (RoPE'd)
  _Float16* khp   = (_Float16*)(ws + 40 * MB);   //  8 MB: K heads f16 (RoPE'd)
  uint32_t* vtp   = (uint32_t*)(ws + 48 * MB);   //  8 MB: V transposed/packed
  _Float16* attnh = (_Float16*)(ws + 56 * MB);   //  8 MB: attention output f16

  k_cast_x<<<16384, 256, 0, stream>>>(x, xh, 4194304);
  k_pack_w<<<2048, 256, 0, stream>>>(Wq, wpq);
  k_pack_w<<<2048, 256, 0, stream>>>(Wk, wpk);
  k_pack_w<<<2048, 256, 0, stream>>>(Wv, wpv);
  k_pack_w<<<2048, 256, 0, stream>>>(Wo, wpo);

  dim3 gg(8, 32, 1);  // N/128, M/128
  k_gemm<<<gg, 256, 0, stream>>>(xh, wpq, pf32, 1024, 1024);
  k_rope_heads<<<8192, 256, 0, stream>>>(pf32, qhp);
  k_gemm<<<gg, 256, 0, stream>>>(xh, wpk, pf32, 1024, 1024);
  k_rope_heads<<<8192, 256, 0, stream>>>(pf32, khp);
  k_gemm<<<gg, 256, 0, stream>>>(xh, wpv, pf32, 1024, 1024);
  k_v_heads_t<<<8192, 256, 0, stream>>>(pf32, vtp);

  k_attn<<<512, 256, 0, stream>>>(qhp, khp, vtp, am, attnh);

  k_gemm<<<gg, 256, 0, stream>>>(attnh, wpo, out, 1024, 1024);
}
